// NMPEdge_59450937311947
// MI455X (gfx1250) — matrix operand order repaired and checked by reading, not compiled, not run on hardware
//
#include <hip/hip_runtime.h>

#define HD 128
#define NG 150
#define NGP 160          // gaussians padded to multiple of 32
#define SP  136          // padded LDS row stride for HD-wide tiles (+8 halfs = 4 banks)
#define SH  264          // padded LDS row stride for 256-wide hidden tiles
#define LOG2F_ 0.69314718055994530942f

typedef __attribute__((ext_vector_type(16))) _Float16 v16h;
typedef __attribute__((ext_vector_type(8)))  _Float16 v8h;
typedef __attribute__((ext_vector_type(8)))  float    v8f;

// ---------------------------------------------------------------- helpers
__device__ __forceinline__ float ssp_f(float x) {
  // shifted softplus: log(1+exp(x)) - log(2), stable for large x
  float sp = (x > 20.0f) ? x : __logf(1.0f + __expf(x));
  return sp - LOG2F_;
}

// A fragment: 16x32 f16 tile, row-major with stride lda (ISA 7.12.2 layout)
// -> 2x ds_load_b128 per lane; lda padded so rows land on distinct banks
__device__ __forceinline__ v16h ldsA(const _Float16* A, int lda, int lane) {
  int row = lane & 15;
  int kb  = (lane < 16) ? 0 : 8;
  const _Float16* p = A + row * lda;
  v16h a;
#pragma unroll
  for (int i = 0; i < 8; ++i) {
    a[i]     = p[kb + i];
    a[i + 8] = p[16 + kb + i];
  }
  return a;
}

// B fragment from pre-packed (fragment-major) weights: lane's 16 halfs contiguous.
__device__ __forceinline__ v16h ldBpk(const _Float16* Bpk, int tile, int lane) {
  return *(const v16h*)(Bpk + (((size_t)tile * 32 + lane) << 4));
}

__device__ __forceinline__ v8f wmma16(v16h a, v16h b, v8f c) {
  return __builtin_amdgcn_wmma_f32_16x16x32_f16(false, a, false, b,
                                                (short)0, c, false, false);
}

// ---------------------------------------------------------------- utility kernels
__global__ void k_zero_f32(float* p, int n) {
  int i = blockIdx.x * blockDim.x + threadIdx.x;
  if (i < n) p[i] = 0.0f;
}

// Pack f32 weight [srcRows x N] (row-major, zero-pad K to Kpad) into
// WMMA-B fragment-major f16: dst[((kt*NT+nt)*32+lane)*16+e]
__global__ void k_pack_b(const float* __restrict__ src, _Float16* __restrict__ dst,
                         int srcRows, int Kpad, int N) {
  int i = blockIdx.x * blockDim.x + threadIdx.x;
  int total = Kpad * N;
  if (i >= total) return;
  int NT   = N >> 4;
  int e    = i & 15;
  int lane = (i >> 4) & 31;
  int tile = i >> 9;              // kt*NT + nt
  int nt = tile % NT;
  int kt = tile / NT;
  int K   = kt * 32 + ((e & 8) ? 16 : 0) + ((lane & 16) ? 8 : 0) + (e & 7);
  int col = nt * 16 + (lane & 15);
  dst[i] = (K < srcRows) ? (_Float16)src[K * N + col] : (_Float16)0.0f;
}

__global__ void k_embed(const int* __restrict__ z, const float* __restrict__ emb,
                        float* __restrict__ h32, _Float16* __restrict__ h16, int n) {
  int i = blockIdx.x * blockDim.x + threadIdx.x;     // one thread per 8 channels
  if (i >= n * (HD / 8)) return;
  int node = i >> 4, c = (i & 15) << 3;
  v8f v = *(const v8f*)(emb + (size_t)z[node] * HD + c);
  *(v8f*)(h32 + (size_t)node * HD + c) = v;
  v8h hv;
#pragma unroll
  for (int j = 0; j < 8; ++j) hv[j] = (_Float16)v[j];
  *(v8h*)(h16 + (size_t)node * HD + c) = hv;
}

__global__ void k_dist(const float* __restrict__ pos,
                       const int* __restrict__ rowIdx, const int* __restrict__ colIdx,
                       float* __restrict__ dist, int E_) {
  int e = blockIdx.x * blockDim.x + threadIdx.x;
  if (e >= E_) return;
  int a = rowIdx[e], b = colIdx[e];
  float dx = pos[a * 3 + 0] - pos[b * 3 + 0];
  float dy = pos[a * 3 + 1] - pos[b * 3 + 1];
  float dz = pos[a * 3 + 2] - pos[b * 3 + 2];
  dist[e] = sqrtf(dx * dx + dy * dy + dz * dz);
}

// one thread per 8-gaussian chunk -> coalesced 16B stores
__global__ void k_gauss(const float* __restrict__ dist, _Float16* __restrict__ g, int E_) {
  int i = blockIdx.x * blockDim.x + threadIdx.x;
  int total = E_ * (NGP / 8);
  if (i >= total) return;
  int e  = i / (NGP / 8);
  int k0 = (i - e * (NGP / 8)) * 8;
  float d = dist[e];
  const float step  = 14.9f / 149.0f;                 // 0.1
  const float coeff = -0.5f / (step * step);          // -50
  v8h out;
#pragma unroll
  for (int j = 0; j < 8; ++j) {
    int k = k0 + j;
    float val = 0.0f;
    if (k < NG) {
      float t = d - step * (float)k;
      val = __expf(coeff * t * t);
    }
    out[j] = (_Float16)val;
  }
  *(v8h*)(g + (size_t)e * NGP + k0) = out;
}

// ---------------------------------------------------------------- EdgeUpdate
// per block: 64 edges; cat = [h[col] | h[row] | efeat]  (K1 = 256+efdim, mult of 32)
// e_out = ssp(cat @ w1 + b1) @ w2 + b2   -> f16 [E, 128]
// each wave: 4 M-tiles share every B fragment (4 WMMA per global B load)
__global__ void __launch_bounds__(256)
k_edge_update(const _Float16* __restrict__ h16,
              const int* __restrict__ rowIdx, const int* __restrict__ colIdx,
              const _Float16* __restrict__ efeat, int efdim, int K1,
              const _Float16* __restrict__ w1pk, const float* __restrict__ b1,
              const _Float16* __restrict__ w2pk, const float* __restrict__ b2,
              _Float16* __restrict__ eout) {
  __shared__ _Float16 sCat[64 * 424];   // padded stride; reused as out staging [64 x SP]
  __shared__ _Float16 sHid[64 * SH];
  __shared__ int sRow[64];
  __shared__ int sCol[64];
  const int tid  = threadIdx.x;
  const int lane = tid & 31;
  const int wave = tid >> 5;
  const int base = blockIdx.x * 64;
  const int PK1  = K1 + 8;              // padded LDS stride (424 or 392)

  if (tid < 64) {
    sRow[tid] = rowIdx[base + tid];
    sCol[tid] = colIdx[base + tid];
  }
  __syncthreads();

  // phase 1: gather concat tile [64 x K1] into LDS, 8 halfs per op
  {
    const int K8 = K1 >> 3;
    for (int idx = tid; idx < 64 * K8; idx += 256) {
      int er = idx / K8;
      int c  = (idx - er * K8) << 3;
      const _Float16* srcp;
      if (c < HD)           srcp = h16 + (size_t)sCol[er] * HD + c;
      else if (c < 2 * HD)  srcp = h16 + (size_t)sRow[er] * HD + (c - HD);
      else                  srcp = efeat + (size_t)(base + er) * efdim + (c - 2 * HD);
      *(v8h*)(sCat + er * PK1 + c) = *(const v8h*)srcp;
    }
  }
  __syncthreads();

  // phase 2: hidden = ssp(cat @ w1 + b1)   [64 x 256]
  for (int nt = wave; nt < 16; nt += 8) {
    v8f acc0 = {}, acc1 = {}, acc2 = {}, acc3 = {};
    for (int kk = 0; kk < K1; kk += 32) {
      v16h b = ldBpk(w1pk, (kk >> 5) * 16 + nt, lane);
      acc0 = wmma16(ldsA(sCat + kk, PK1, lane), b, acc0);
      acc1 = wmma16(ldsA(sCat + 16 * PK1 + kk, PK1, lane), b, acc1);
      acc2 = wmma16(ldsA(sCat + 32 * PK1 + kk, PK1, lane), b, acc2);
      acc3 = wmma16(ldsA(sCat + 48 * PK1 + kk, PK1, lane), b, acc3);
    }
    int cc = nt * 16 + (lane & 15);
    float bias = b1[cc];
    int rbase = (lane < 16) ? 0 : 8;
#pragma unroll
    for (int r = 0; r < 8; ++r) {
      sHid[(rbase + r) * SH + cc]      = (_Float16)ssp_f(acc0[r] + bias);
      sHid[(16 + rbase + r) * SH + cc] = (_Float16)ssp_f(acc1[r] + bias);
      sHid[(32 + rbase + r) * SH + cc] = (_Float16)ssp_f(acc2[r] + bias);
      sHid[(48 + rbase + r) * SH + cc] = (_Float16)ssp_f(acc3[r] + bias);
    }
  }
  __syncthreads();

  // phase 3: out = hidden @ w2 + b2   [64 x 128], staged in LDS
  {
    int nt = wave;
    v8f acc0 = {}, acc1 = {}, acc2 = {}, acc3 = {};
    for (int kk = 0; kk < 256; kk += 32) {
      v16h b = ldBpk(w2pk, (kk >> 5) * 8 + nt, lane);
      acc0 = wmma16(ldsA(sHid + kk, SH, lane), b, acc0);
      acc1 = wmma16(ldsA(sHid + 16 * SH + kk, SH, lane), b, acc1);
      acc2 = wmma16(ldsA(sHid + 32 * SH + kk, SH, lane), b, acc2);
      acc3 = wmma16(ldsA(sHid + 48 * SH + kk, SH, lane), b, acc3);
    }
    int cc = nt * 16 + (lane & 15);
    float bias = b2[cc];
    int rbase = (lane < 16) ? 0 : 8;
#pragma unroll
    for (int r = 0; r < 8; ++r) {
      sCat[(rbase + r) * SP + cc]      = (_Float16)(acc0[r] + bias);
      sCat[(16 + rbase + r) * SP + cc] = (_Float16)(acc1[r] + bias);
      sCat[(32 + rbase + r) * SP + cc] = (_Float16)(acc2[r] + bias);
      sCat[(48 + rbase + r) * SP + cc] = (_Float16)(acc3[r] + bias);
    }
  }
  __syncthreads();

  // phase 4: coalesced store of eout tile (64*128/8 = 1024 chunks)
  for (int idx = tid; idx < 1024; idx += 256) {
    int er = idx >> 4, c = (idx & 15) << 3;
    *(v8h*)(eout + (size_t)(base + er) * HD + c) = *(const v8h*)(sCat + er * SP + c);
  }
}

// ---------------------------------------------------------------- CFConv
// W = ssp(ssp(e@fw1+fb1)@fw2+fb2); msg = (h[row]@cfw)*W; m[col] += msg
__global__ void __launch_bounds__(256)
k_cfconv(const _Float16* __restrict__ h16,
         const int* __restrict__ rowIdx, const int* __restrict__ colIdx,
         const _Float16* __restrict__ e16,
         const _Float16* __restrict__ fw1, const float* __restrict__ fb1,
         const _Float16* __restrict__ fw2, const float* __restrict__ fb2,
         const _Float16* __restrict__ cfw,
         float* __restrict__ m) {
  __shared__ _Float16 sE[64 * SP];
  __shared__ _Float16 sHr[64 * SP];
  __shared__ _Float16 sT1[64 * SP];
  __shared__ _Float16 sWf[64 * SP];
  __shared__ int sRow[64];
  __shared__ int sCol[64];
  const int tid  = threadIdx.x;
  const int lane = tid & 31;
  const int wave = tid >> 5;
  const int base = blockIdx.x * 64;

  if (tid < 64) {
    sRow[tid] = rowIdx[base + tid];
    sCol[tid] = colIdx[base + tid];
  }
  __syncthreads();

  for (int idx = tid; idx < 1024; idx += 256) {
    int er = idx >> 4, c = (idx & 15) << 3;
    *(v8h*)(sE + er * SP + c)  = *(const v8h*)(e16 + (size_t)(base + er) * HD + c);
    *(v8h*)(sHr + er * SP + c) = *(const v8h*)(h16 + (size_t)sRow[er] * HD + c);
  }
  __syncthreads();

  const int nt    = wave;
  const int cc    = nt * 16 + (lane & 15);
  const int rbase = (lane < 16) ? 0 : 8;

  // t1 = ssp(e @ fw1 + fb1)
  {
    v8f acc0 = {}, acc1 = {}, acc2 = {}, acc3 = {};
    for (int kk = 0; kk < HD; kk += 32) {
      v16h b = ldBpk(fw1, (kk >> 5) * 8 + nt, lane);
      acc0 = wmma16(ldsA(sE + kk, SP, lane), b, acc0);
      acc1 = wmma16(ldsA(sE + 16 * SP + kk, SP, lane), b, acc1);
      acc2 = wmma16(ldsA(sE + 32 * SP + kk, SP, lane), b, acc2);
      acc3 = wmma16(ldsA(sE + 48 * SP + kk, SP, lane), b, acc3);
    }
    float bias = fb1[cc];
#pragma unroll
    for (int r = 0; r < 8; ++r) {
      sT1[(rbase + r) * SP + cc]      = (_Float16)ssp_f(acc0[r] + bias);
      sT1[(16 + rbase + r) * SP + cc] = (_Float16)ssp_f(acc1[r] + bias);
      sT1[(32 + rbase + r) * SP + cc] = (_Float16)ssp_f(acc2[r] + bias);
      sT1[(48 + rbase + r) * SP + cc] = (_Float16)ssp_f(acc3[r] + bias);
    }
  }
  __syncthreads();

  // wf = ssp(t1 @ fw2 + fb2), staged to LDS (each wave writes/reads its own cols)
  {
    v8f acc0 = {}, acc1 = {}, acc2 = {}, acc3 = {};
    for (int kk = 0; kk < HD; kk += 32) {
      v16h b = ldBpk(fw2, (kk >> 5) * 8 + nt, lane);
      acc0 = wmma16(ldsA(sT1 + kk, SP, lane), b, acc0);
      acc1 = wmma16(ldsA(sT1 + 16 * SP + kk, SP, lane), b, acc1);
      acc2 = wmma16(ldsA(sT1 + 32 * SP + kk, SP, lane), b, acc2);
      acc3 = wmma16(ldsA(sT1 + 48 * SP + kk, SP, lane), b, acc3);
    }
    float bias = fb2[cc];
#pragma unroll
    for (int r = 0; r < 8; ++r) {
      sWf[(rbase + r) * SP + cc]      = (_Float16)ssp_f(acc0[r] + bias);
      sWf[(16 + rbase + r) * SP + cc] = (_Float16)ssp_f(acc1[r] + bias);
      sWf[(32 + rbase + r) * SP + cc] = (_Float16)ssp_f(acc2[r] + bias);
      sWf[(48 + rbase + r) * SP + cc] = (_Float16)ssp_f(acc3[r] + bias);
    }
  }

  // hm = h[row] @ cfw;  msg = hm * wf;  scatter-add to m[col]
  {
    v8f acc0 = {}, acc1 = {}, acc2 = {}, acc3 = {};
    for (int kk = 0; kk < HD; kk += 32) {
      v16h b = ldBpk(cfw, (kk >> 5) * 8 + nt, lane);
      acc0 = wmma16(ldsA(sHr + kk, SP, lane), b, acc0);
      acc1 = wmma16(ldsA(sHr + 16 * SP + kk, SP, lane), b, acc1);
      acc2 = wmma16(ldsA(sHr + 32 * SP + kk, SP, lane), b, acc2);
      acc3 = wmma16(ldsA(sHr + 48 * SP + kk, SP, lane), b, acc3);
    }
#pragma unroll
    for (int r = 0; r < 8; ++r) {
      int r0 = rbase + r;
      atomicAdd(&m[(size_t)sCol[r0] * HD + cc],
                acc0[r] * (float)sWf[r0 * SP + cc]);
      atomicAdd(&m[(size_t)sCol[16 + r0] * HD + cc],
                acc1[r] * (float)sWf[(16 + r0) * SP + cc]);
      atomicAdd(&m[(size_t)sCol[32 + r0] * HD + cc],
                acc2[r] * (float)sWf[(32 + r0) * SP + cc]);
      atomicAdd(&m[(size_t)sCol[48 + r0] * HD + cc],
                acc3[r] * (float)sWf[(48 + r0) * SP + cc]);
    }
  }
}

// ---------------------------------------------------------------- StateMLP
// h += ssp(m @ st_w1) @ st_w2   (no biases); 32 nodes per block
__global__ void __launch_bounds__(256)
k_state(const float* __restrict__ m,
        const _Float16* __restrict__ sw1, const _Float16* __restrict__ sw2,
        float* __restrict__ h32, _Float16* __restrict__ h16) {
  __shared__ _Float16 sM[32 * SP];
  __shared__ _Float16 sT[32 * SP];
  __shared__ float    sO[32 * SP];
  const int tid  = threadIdx.x;
  const int lane = tid & 31;
  const int wave = tid >> 5;
  const int base = blockIdx.x * 32;

  for (int idx = tid; idx < 512; idx += 256) {
    int er = idx >> 4, c = (idx & 15) << 3;
    v8f mv = *(const v8f*)(m + (size_t)(base + er) * HD + c);
    v8h hv;
#pragma unroll
    for (int j = 0; j < 8; ++j) hv[j] = (_Float16)mv[j];
    *(v8h*)(sM + er * SP + c) = hv;
  }
  __syncthreads();

  const int nt    = wave;
  const int cc    = nt * 16 + (lane & 15);
  const int rbase = (lane < 16) ? 0 : 8;

  {
    v8f acc0 = {}, acc1 = {};
    for (int kk = 0; kk < HD; kk += 32) {
      v16h b = ldBpk(sw1, (kk >> 5) * 8 + nt, lane);
      acc0 = wmma16(ldsA(sM + kk, SP, lane), b, acc0);
      acc1 = wmma16(ldsA(sM + 16 * SP + kk, SP, lane), b, acc1);
    }
#pragma unroll
    for (int r = 0; r < 8; ++r) {
      sT[(rbase + r) * SP + cc]      = (_Float16)ssp_f(acc0[r]);
      sT[(16 + rbase + r) * SP + cc] = (_Float16)ssp_f(acc1[r]);
    }
  }
  __syncthreads();
  {
    v8f acc0 = {}, acc1 = {};
    for (int kk = 0; kk < HD; kk += 32) {
      v16h b = ldBpk(sw2, (kk >> 5) * 8 + nt, lane);
      acc0 = wmma16(ldsA(sT + kk, SP, lane), b, acc0);
      acc1 = wmma16(ldsA(sT + 16 * SP + kk, SP, lane), b, acc1);
    }
#pragma unroll
    for (int r = 0; r < 8; ++r) {
      sO[(rbase + r) * SP + cc]      = acc0[r];
      sO[(16 + rbase + r) * SP + cc] = acc1[r];
    }
  }
  __syncthreads();

  // coalesced residual update
  for (int idx = tid; idx < 512; idx += 256) {
    int er = idx >> 4, c = (idx & 15) << 3;
    size_t g = (size_t)(base + er) * HD + c;
    v8f hv = *(const v8f*)(h32 + g);
    v8f dv = *(const v8f*)(sO + er * SP + c);
    v8h h16v;
#pragma unroll
    for (int j = 0; j < 8; ++j) {
      hv[j] += dv[j];
      h16v[j] = (_Float16)hv[j];
    }
    *(v8f*)(h32 + g) = hv;
    *(v8h*)(h16 + g) = h16v;
  }
}

// ---------------------------------------------------------------- readout
__global__ void k_readout(const float* __restrict__ h32, const int* __restrict__ batch,
                          const float* __restrict__ w1, const float* __restrict__ b1,
                          const float* __restrict__ w2, const float* __restrict__ b2,
                          float* __restrict__ out, int n) {
  int node = blockIdx.x * blockDim.x + threadIdx.x;
  if (node >= n) return;
  const float* hr = h32 + (size_t)node * HD;
  float val = b2[0];
  for (int j = 0; j < 64; ++j) {
    float s = b1[j];
    for (int c = 0; c < HD; ++c) s += hr[c] * w1[c * 64 + j];
    val += ssp_f(s) * w2[j];
  }
  atomicAdd(&out[batch[node]], val);
}

// ---------------------------------------------------------------- launch
extern "C" void kernel_launch(void* const* d_in, const int* in_sizes, int n_in,
                              void* d_out, int out_size, void* d_ws, size_t ws_size,
                              hipStream_t stream) {
  (void)in_sizes; (void)n_in; (void)out_size; (void)ws_size;
  const int N_ = 20000, E_ = 256000, L_ = 3, B_ = 16;

  const int*   z      = (const int*)d_in[0];
  const float* pos    = (const float*)d_in[1];
  const int*   batch  = (const int*)d_in[2];
  const int*   eidx   = (const int*)d_in[3];
  const int*   rowIdx = eidx;        // edge_index[0] (source j)
  const int*   colIdx = eidx + E_;   // edge_index[1] (target i)
  const float* emb    = (const float*)d_in[4];
  const float* eu0_w1 = (const float*)d_in[5];   // [406,256]
  const float* eu_w1  = (const float*)d_in[6];   // [2,384,256]
  const float* eu_b1  = (const float*)d_in[7];   // [3,256]
  const float* eu_w2  = (const float*)d_in[8];   // [3,256,128]
  const float* eu_b2  = (const float*)d_in[9];   // [3,128]
  const float* flt_w1 = (const float*)d_in[10];
  const float* flt_b1 = (const float*)d_in[11];
  const float* flt_w2 = (const float*)d_in[12];
  const float* flt_b2 = (const float*)d_in[13];
  const float* cf_w   = (const float*)d_in[14];
  const float* st_w1  = (const float*)d_in[15];
  const float* st_w2  = (const float*)d_in[16];
  const float* o_w1   = (const float*)d_in[17];
  const float* o_b1   = (const float*)d_in[18];
  const float* o_w2   = (const float*)d_in[19];
  const float* o_b2   = (const float*)d_in[20];

  char* wsb = (char*)d_ws;
  size_t off = 0;
  auto alloc = [&](size_t bytes) -> void* {
    void* p = wsb + off;
    off = (off + bytes + 255) & ~(size_t)255;
    return p;
  };
  float*    h32  = (float*)alloc((size_t)N_ * HD * 4);
  _Float16* h16  = (_Float16*)alloc((size_t)N_ * HD * 2);
  _Float16* gaus = (_Float16*)alloc((size_t)E_ * NGP * 2);
  _Float16* e16  = (_Float16*)alloc((size_t)E_ * HD * 2);
  float*    msum = (float*)alloc((size_t)N_ * HD * 4);
  float*    dbuf = (float*)alloc((size_t)E_ * 4);
  _Float16* w0   = (_Float16*)alloc((size_t)416 * 256 * 2);
  _Float16* w12  = (_Float16*)alloc((size_t)2 * 384 * 256 * 2);
  _Float16* wfc2 = (_Float16*)alloc((size_t)3 * 256 * 128 * 2);
  _Float16* wf1  = (_Float16*)alloc((size_t)3 * 128 * 128 * 2);
  _Float16* wf2  = (_Float16*)alloc((size_t)3 * 128 * 128 * 2);
  _Float16* wcf  = (_Float16*)alloc((size_t)3 * 128 * 128 * 2);
  _Float16* ws1  = (_Float16*)alloc((size_t)3 * 128 * 128 * 2);
  _Float16* ws2  = (_Float16*)alloc((size_t)3 * 128 * 128 * 2);

  // fragment-major weight packing (zero-padded in K where needed)
  auto pack = [&](const float* s, _Float16* d, int sr, int Kpad, int N) {
    int total = Kpad * N;
    k_pack_b<<<(total + 255) / 256, 256, 0, stream>>>(s, d, sr, Kpad, N);
  };
  pack(eu0_w1, w0, 406, 416, 256);                       // K 406 -> 416 pad
  for (int i = 0; i < 2; ++i)
    pack(eu_w1 + (size_t)i * 384 * 256, w12 + (size_t)i * 384 * 256, 384, 384, 256);
  for (int i = 0; i < L_; ++i) {
    pack(eu_w2  + (size_t)i * 256 * 128, wfc2 + (size_t)i * 256 * 128, 256, 256, 128);
    pack(flt_w1 + (size_t)i * 128 * 128, wf1  + (size_t)i * 128 * 128, 128, 128, 128);
    pack(flt_w2 + (size_t)i * 128 * 128, wf2  + (size_t)i * 128 * 128, 128, 128, 128);
    pack(cf_w   + (size_t)i * 128 * 128, wcf  + (size_t)i * 128 * 128, 128, 128, 128);
    pack(st_w1  + (size_t)i * 128 * 128, ws1  + (size_t)i * 128 * 128, 128, 128, 128);
    pack(st_w2  + (size_t)i * 128 * 128, ws2  + (size_t)i * 128 * 128, 128, 128, 128);
  }

  k_zero_f32<<<1, 32, 0, stream>>>((float*)d_out, B_);
  k_embed<<<(N_ * (HD / 8) + 255) / 256, 256, 0, stream>>>(z, emb, h32, h16, N_);
  k_dist<<<(E_ + 255) / 256, 256, 0, stream>>>(pos, rowIdx, colIdx, dbuf, E_);
  k_gauss<<<(E_ * (NGP / 8) + 255) / 256, 256, 0, stream>>>(dbuf, gaus, E_);

  for (int i = 0; i < L_; ++i) {
    const _Float16* ef = (i == 0) ? gaus : e16;
    int efdim = (i == 0) ? NGP : HD;
    int K1 = 2 * HD + efdim;                  // 416 or 384 (multiples of 32)
    const _Float16* w1p = (i == 0) ? w0 : (w12 + (size_t)(i - 1) * 384 * 256);

    k_edge_update<<<E_ / 64, 256, 0, stream>>>(
        h16, rowIdx, colIdx, ef, efdim, K1,
        w1p, eu_b1 + i * 256,
        wfc2 + (size_t)i * 256 * 128, eu_b2 + i * 128, e16);

    k_zero_f32<<<(N_ * HD + 255) / 256, 256, 0, stream>>>(msum, N_ * HD);

    k_cfconv<<<E_ / 64, 256, 0, stream>>>(
        h16, rowIdx, colIdx, e16,
        wf1 + (size_t)i * 128 * 128, flt_b1 + i * 128,
        wf2 + (size_t)i * 128 * 128, flt_b2 + i * 128,
        wcf + (size_t)i * 128 * 128, msum);

    k_state<<<N_ / 32, 256, 0, stream>>>(
        msum, ws1 + (size_t)i * 128 * 128, ws2 + (size_t)i * 128 * 128, h32, h16);
  }

  k_readout<<<(N_ + 127) / 128, 128, 0, stream>>>(
      h32, batch, o_w1, o_b1, o_w2, o_b2, (float*)d_out, N_);
}
